// Node_cls_Module_75788992905498
// MI455X (gfx1250) — matrix operand order repaired and checked by reading, not compiled, not run on hardware
//
#include <hip/hip_runtime.h>
#include <hip/hip_bf16.h>
#include <stdint.h>

// ---- problem constants (match reference) ----
#define NNODES   50000
#define FIN      128
#define HIDC     64      // HID == OUT == 64
#define NHEAD    2
#define NCLS     10
#define NEG_SLOPE 0.2f

typedef __attribute__((ext_vector_type(16))) _Float16 v16h;
typedef __attribute__((ext_vector_type(8)))  _Float16 v8h;
typedef __attribute__((ext_vector_type(8)))  float    v8f;

static __device__ __forceinline__ float leaky(float v) {
  return v > 0.f ? v : NEG_SLOPE * v;
}
// monotone float -> uint key for atomicMax on floats
static __device__ __forceinline__ unsigned fkey(float f) {
  unsigned u = __float_as_uint(f);
  return (u & 0x80000000u) ? ~u : (u | 0x80000000u);
}
static __device__ __forceinline__ float funkey(unsigned k) {
  return __uint_as_float((k & 0x80000000u) ? (k & 0x7FFFFFFFu) : ~k);
}

// ---------------- utility kernels ----------------
__global__ void gat_zero_f32(float* __restrict__ p, int n) {
  int t = blockIdx.x * blockDim.x + threadIdx.x;
  if (t < n) p[t] = 0.f;
}

__global__ void gat_f32_to_f16(const float* __restrict__ in, _Float16* __restrict__ out, int n) {
  int t = blockIdx.x * blockDim.x + threadIdx.x;
  if (t < n) out[t] = (_Float16)in[t];
}

// Pack row-major W[K][Nn] (f32) into WMMA B-fragment layout (f16):
// Bp[(((nt*ktiles)+kt)*32 + lane)*16 + j] = W[kt*32 + (lane<16?0:16) + j][nt*16 + (lane&15)]
__global__ void gat_pack_b(const float* __restrict__ W, _Float16* __restrict__ Bp, int K, int Nn) {
  int idx = blockIdx.x * blockDim.x + threadIdx.x;
  int ktiles = K >> 5;
  int ntiles = Nn >> 4;
  int total = ntiles * ktiles * 32 * 16;
  if (idx >= total) return;
  int j    = idx & 15;
  int lane = (idx >> 4) & 31;
  int t    = idx >> 9;
  int kt   = t % ktiles;
  int nt   = t / ktiles;
  int n = nt * 16 + (lane & 15);
  int k = kt * 32 + ((lane < 16) ? 0 : 16) + j;
  Bp[idx] = (_Float16)W[(size_t)k * Nn + n];
}

// ---------------- WMMA GEMM: C[M,Nn] = A[M,K](f16 row-major) x Bp(packed f16) ----------------
// one wave (32 lanes) computes one 16x16 output tile, K accumulated in steps of 32
__global__ void gat_wmma_gemm(const _Float16* __restrict__ A,
                              const _Float16* __restrict__ Bp,
                              float* __restrict__ C, int M, int K, int Nn) {
  const int lane = threadIdx.x & 31;
  const int wave = (blockIdx.x * blockDim.x + threadIdx.x) >> 5;
  const int ntiles = Nn >> 4;
  const int mt = wave / ntiles;
  const int nt = wave - mt * ntiles;
  if (mt * 16 >= M) return;
  const int ktiles = K >> 5;

  // A fragment: 16x32 f16. lanes 0-15: row M=lane, halfs = K{0..15};
  // lanes 16-31: row M=lane-16, halfs = K{16..31}
  const int rowA = mt * 16 + (lane & 15);
  const int koff = (lane < 16) ? 0 : 16;
  const _Float16* Ap = A + (size_t)rowA * K + koff;
  // B fragment: packed contiguously, 512 halfs per K-tile
  const _Float16* Bq = Bp + ((size_t)(nt * ktiles) * 32 + lane) * 16;

  v8f acc = {};
  for (int kt = 0; kt < ktiles; ++kt) {
    v8h a0 = *(const v8h*)(Ap);
    v8h a1 = *(const v8h*)(Ap + 8);
    v16h a = __builtin_shufflevector(a0, a1, 0,1,2,3,4,5,6,7,8,9,10,11,12,13,14,15);
    v16h b = *(const v16h*)(Bq);
    acc = __builtin_amdgcn_wmma_f32_16x16x32_f16(false, a, false, b, (short)0, acc, false, false);
    Ap += 32;
    Bq += 512;
  }

  // C/D layout: lane 0-15 -> M=0..7 in acc[0..7], lane 16-31 -> M=8..15; N = lane&15
  float* Cp = C + (size_t)mt * 16 * Nn + nt * 16 + (lane & 15);
  const int mbase = (lane < 16) ? 0 : 8;
#pragma unroll
  for (int r = 0; r < 8; ++r)
    Cp[(size_t)(mbase + r) * Nn] = acc[r];
}

// ---------------- attention kernels ----------------
// Hm: [N, NHEAD*C] with head h channel c at column h*C+c
__global__ void gat_att_logits(const float* __restrict__ Hm,
                               const float* __restrict__ a_src, const float* __restrict__ a_dst,
                               float* __restrict__ ALS, float* __restrict__ ALD, int Nn, int C) {
  int t = blockIdx.x * blockDim.x + threadIdx.x;
  if (t >= Nn * NHEAD) return;
  int n = t / NHEAD, h = t - n * NHEAD;
  const float* hp = Hm + (size_t)n * (NHEAD * C) + h * C;
  const float* as = a_src + h * C;
  const float* ad = a_dst + h * C;
  float s = 0.f, d = 0.f;
  for (int c = 0; c < C; ++c) { float v = hp[c]; s += v * as[c]; d += v * ad[c]; }
  ALS[t] = s;
  ALD[t] = d;
}

__global__ void gat_edge_max(const long long* __restrict__ ei,
                             const float* __restrict__ ALS, const float* __restrict__ ALD,
                             unsigned* __restrict__ MX, int E) {
  int t = blockIdx.x * blockDim.x + threadIdx.x;
  if (t >= E * NHEAD) return;
  int e = t >> 1, h = t & 1;
  int s = (int)ei[e];
  int d = (int)ei[(size_t)E + e];
  float v = leaky(ALS[s * NHEAD + h] + ALD[d * NHEAD + h]);
  atomicMax(&MX[d * NHEAD + h], fkey(v));
}

__global__ void gat_edge_exp(const long long* __restrict__ ei,
                             const float* __restrict__ ALS, const float* __restrict__ ALD,
                             const unsigned* __restrict__ MX,
                             float* __restrict__ P, float* __restrict__ Z, int E) {
  int t = blockIdx.x * blockDim.x + threadIdx.x;
  if (t >= E * NHEAD) return;
  int e = t >> 1, h = t & 1;
  int s = (int)ei[e];
  int d = (int)ei[(size_t)E + e];
  float v = leaky(ALS[s * NHEAD + h] + ALD[d * NHEAD + h]);
  float m = funkey(MX[d * NHEAD + h]);
  float p = __expf(v - m);
  P[t] = p;
  atomicAdd(&Z[d * NHEAD + h], p);
}

// one thread per (edge, head, 4-channel chunk): float4 gather + 4 float atomics
__global__ void gat_edge_agg(const long long* __restrict__ ei,
                             const float* __restrict__ P, const float* __restrict__ Z,
                             const float* __restrict__ Hm, float* __restrict__ AGG, int E) {
  int t = blockIdx.x * blockDim.x + threadIdx.x;
  if (t >= E * 32) return;               // NHEAD*C/4 = 2*64/4 = 32 chunks per edge
  int e = t >> 5;
  int q = t & 31;
  int h = q >> 4;
  int c = (q & 15) << 2;
  int s = (int)ei[e];
  int d = (int)ei[(size_t)E + e];
  float alpha = P[e * NHEAD + h] / Z[d * NHEAD + h];
  const float4 hv = *(const float4*)(Hm + (size_t)s * (NHEAD * HIDC) + h * HIDC + c);
  float* dp = AGG + (size_t)d * (NHEAD * HIDC) + h * HIDC + c;
  atomicAdd(dp + 0, alpha * hv.x);
  atomicAdd(dp + 1, alpha * hv.y);
  atomicAdd(dp + 2, alpha * hv.z);
  atomicAdd(dp + 3, alpha * hv.w);
}

// head mean + bias (+optional relu); optional f32 and f16 outputs
__global__ void gat_finalize(const float* __restrict__ AGG, const float* __restrict__ bias,
                             float* __restrict__ outf, _Float16* __restrict__ outh,
                             int Nn, int do_relu) {
  int t = blockIdx.x * blockDim.x + threadIdx.x;
  if (t >= Nn * HIDC) return;
  int n = t >> 6, c = t & 63;
  float v = 0.5f * (AGG[(size_t)n * 128 + c] + AGG[(size_t)n * 128 + 64 + c]) + bias[c];
  if (do_relu) v = fmaxf(v, 0.f);
  if (outf) outf[t] = v;
  if (outh) outh[t] = (_Float16)v;
}

__global__ void gat_classifier(const float* __restrict__ F2, const float* __restrict__ Wc,
                               const float* __restrict__ bc, float* __restrict__ preds, int Nn) {
  int t = blockIdx.x * blockDim.x + threadIdx.x;
  if (t >= Nn * NCLS) return;
  int n = t / NCLS, k = t - n * NCLS;
  const float* f = F2 + (size_t)n * HIDC;
  float acc = bc[k];
  for (int c = 0; c < HIDC; ++c) acc += f[c] * Wc[c * NCLS + k];
  preds[t] = acc;
}

__global__ void gat_copy_y(const long long* __restrict__ y, long long* __restrict__ out, int Nn) {
  int t = blockIdx.x * blockDim.x + threadIdx.x;
  if (t < Nn) out[t] = y[t];
}

#define CDIV(a, b) (((a) + (b) - 1) / (b))

extern "C" void kernel_launch(void* const* d_in, const int* in_sizes, int n_in,
                              void* d_out, int out_size, void* d_ws, size_t ws_size,
                              hipStream_t stream) {
  const float*     x   = (const float*)d_in[0];
  const long long* ei  = (const long long*)d_in[1];
  const long long* y   = (const long long*)d_in[3];
  const float*     W1  = (const float*)d_in[4];
  const float*     as1 = (const float*)d_in[5];
  const float*     ad1 = (const float*)d_in[6];
  const float*     b1  = (const float*)d_in[7];
  const float*     W2  = (const float*)d_in[8];
  const float*     as2 = (const float*)d_in[9];
  const float*     ad2 = (const float*)d_in[10];
  const float*     b2  = (const float*)d_in[11];
  const float*     Wc  = (const float*)d_in[12];
  const float*     bc  = (const float*)d_in[13];

  const int Nn = NNODES;
  const int E  = in_sizes[1] / 2;

  // ---- workspace carve-out (~85 MB) ----
  char* w = (char*)d_ws;
  size_t off = 0;
  auto carve = [&](size_t bytes) -> void* {
    void* p = w + off;
    off = (off + bytes + 255) & ~(size_t)255;
    return p;
  };
  _Float16* Ah  = (_Float16*)carve((size_t)Nn * FIN * 2);   // f16 GEMM A (x, then h1r)
  _Float16* Bp1 = (_Float16*)carve((size_t)FIN * 128 * 2);  // packed W1
  _Float16* Bp2 = (_Float16*)carve((size_t)HIDC * 128 * 2); // packed W2
  float*    Hm  = (float*)carve((size_t)Nn * 128 * 4);      // GEMM output h [N, H*C]
  float*    ALS = (float*)carve((size_t)Nn * NHEAD * 4);
  float*    ALD = (float*)carve((size_t)Nn * NHEAD * 4);
  unsigned* MX  = (unsigned*)carve((size_t)Nn * NHEAD * 4);
  float*    Z   = (float*)carve((size_t)Nn * NHEAD * 4);
  float*    P   = (float*)carve((size_t)E * NHEAD * 4);
  float*    AGG = (float*)carve((size_t)Nn * 128 * 4);
  float*    F2  = (float*)carve((size_t)Nn * HIDC * 4);

  const int BS = 256;
  float* preds = (float*)d_out;

  // ---- prep: x -> f16, pack weights ----
  gat_f32_to_f16<<<CDIV(Nn * FIN, BS), BS, 0, stream>>>(x, Ah, Nn * FIN);
  gat_pack_b<<<CDIV(FIN * 128, BS), BS, 0, stream>>>(W1, Bp1, FIN, 128);
  gat_pack_b<<<CDIV(HIDC * 128, BS), BS, 0, stream>>>(W2, Bp2, HIDC, 128);

  const int mtiles = Nn / 16;          // 3125 exactly
  const int waves1 = mtiles * (128 / 16);

  // ================= layer 1 =================
  gat_wmma_gemm<<<CDIV(waves1 * 32, BS), BS, 0, stream>>>(Ah, Bp1, Hm, Nn, FIN, 128);
  gat_att_logits<<<CDIV(Nn * NHEAD, BS), BS, 0, stream>>>(Hm, as1, ad1, ALS, ALD, Nn, HIDC);
  gat_zero_f32<<<CDIV(Nn * NHEAD, BS), BS, 0, stream>>>((float*)MX, Nn * NHEAD);
  gat_zero_f32<<<CDIV(Nn * NHEAD, BS), BS, 0, stream>>>(Z, Nn * NHEAD);
  gat_zero_f32<<<CDIV(Nn * 128, BS), BS, 0, stream>>>(AGG, Nn * 128);
  gat_edge_max<<<CDIV(E * NHEAD, BS), BS, 0, stream>>>(ei, ALS, ALD, MX, E);
  gat_edge_exp<<<CDIV(E * NHEAD, BS), BS, 0, stream>>>(ei, ALS, ALD, MX, P, Z, E);
  gat_edge_agg<<<CDIV(E * 32, BS), BS, 0, stream>>>(ei, P, Z, Hm, AGG, E);
  // head mean + b1, relu; emit f16 input for layer-2 GEMM (reuses Ah)
  gat_finalize<<<CDIV(Nn * HIDC, BS), BS, 0, stream>>>(AGG, b1, nullptr, Ah, Nn, 1);

  // ================= layer 2 =================
  gat_wmma_gemm<<<CDIV(waves1 * 32, BS), BS, 0, stream>>>(Ah, Bp2, Hm, Nn, HIDC, 128);
  gat_att_logits<<<CDIV(Nn * NHEAD, BS), BS, 0, stream>>>(Hm, as2, ad2, ALS, ALD, Nn, HIDC);
  gat_zero_f32<<<CDIV(Nn * NHEAD, BS), BS, 0, stream>>>((float*)MX, Nn * NHEAD);
  gat_zero_f32<<<CDIV(Nn * NHEAD, BS), BS, 0, stream>>>(Z, Nn * NHEAD);
  gat_zero_f32<<<CDIV(Nn * 128, BS), BS, 0, stream>>>(AGG, Nn * 128);
  gat_edge_max<<<CDIV(E * NHEAD, BS), BS, 0, stream>>>(ei, ALS, ALD, MX, E);
  gat_edge_exp<<<CDIV(E * NHEAD, BS), BS, 0, stream>>>(ei, ALS, ALD, MX, P, Z, E);
  gat_edge_agg<<<CDIV(E * 32, BS), BS, 0, stream>>>(ei, P, Z, Hm, AGG, E);
  // head mean + b2 (no relu); f32 features for classifier
  gat_finalize<<<CDIV(Nn * HIDC, BS), BS, 0, stream>>>(AGG, b2, F2, nullptr, Nn, 0);

  // ================= classifier + y passthrough =================
  gat_classifier<<<CDIV(Nn * NCLS, BS), BS, 0, stream>>>(F2, Wc, bc, preds, Nn);
  gat_copy_y<<<CDIV(Nn, BS), BS, 0, stream>>>(y, (long long*)(preds + (size_t)Nn * NCLS), Nn);
}
